// PGExplainer_16612933501431
// MI455X (gfx1250) — matrix operand order repaired and checked by reading, not compiled, not run on hardware
//
#include <hip/hip_runtime.h>
#include <math.h>

typedef __attribute__((ext_vector_type(16))) _Float16 v16h;
typedef __attribute__((ext_vector_type(8)))  float    v8f;

#define D_EMB 128
#define K_TOT 384   // 3*D_EMB
#define HID   64
#define NCLS  7

// ---------------- K0: zero workspace words ----------------
__global__ void init_ws_kernel(int* __restrict__ ws, int nwords) {
    int i = blockIdx.x * blockDim.x + threadIdx.x;
    if (i < nwords) ws[i] = 0;
}

// ---------------- K1: scan edges, build incident list + adj row ----------------
__global__ void scan_kernel(const int* __restrict__ row, const int* __restrict__ col,
                            const float* __restrict__ adj_data, const int* __restrict__ nodeid,
                            float* __restrict__ A_cnt, int* __restrict__ counters,
                            int* __restrict__ inc_list, int E) {
    int e = blockIdx.x * blockDim.x + threadIdx.x;
    if (e >= E) return;
    int nid = nodeid[0];
    int r = row[e], c = col[e];
    if (r == nid) atomicAdd(&A_cnt[c], adj_data[e]);      // adj_dense[nid, c] (duplicates sum)
    if (r == nid || c == nid) {
        int slot = atomicAdd(&counters[0], 1);
        inc_list[slot] = e;
    }
}

// ---------------- K2: WMMA MLP -> sigmoid gate for incident edges ----------------
// One wave32 block handles a 16-edge M-tile: h[16,64] = relu(feats[16,384] @ W1 + b1),
// log_alpha = h @ W2 + b2, gate = sigmoid(log_alpha), scattered into Gout/Gin.
__global__ void __launch_bounds__(32) mlp_gate_kernel(
    const float* __restrict__ embed, const int* __restrict__ row, const int* __restrict__ col,
    const float* __restrict__ W1, const float* __restrict__ b1,
    const float* __restrict__ W2, const float* __restrict__ b2,
    const int* __restrict__ nodeid, const int* __restrict__ counters,
    const int* __restrict__ inc_list,
    float* __restrict__ Gout, float* __restrict__ Gin)
{
    __shared__ _Float16 sF[16][K_TOT];   // feats tile, f16
    __shared__ float    sP[32][8];       // per-lane partial row sums
    __shared__ int      sR[16], sC[16];

    const int count = counters[0];
    const int m0 = blockIdx.x * 16;
    if (m0 >= count) return;             // uniform exit; almost all blocks leave here

    const int tid = threadIdx.x;
    const int nid = nodeid[0];

    if (tid < 16) {
        int gi = m0 + tid;
        int e  = (gi < count) ? inc_list[gi] : inc_list[m0];  // pad with a valid edge
        sR[tid] = row[e];
        sC[tid] = col[e];
    }
    __syncthreads();

    // Stage feats = [embed[row] | embed[col] | embed[nid]] as f16 in LDS
    for (int idx = tid; idx < 16 * K_TOT; idx += 32) {
        int i = idx / K_TOT;
        int k = idx - i * K_TOT;
        float v;
        if (k < D_EMB)          v = embed[(long)sR[i] * D_EMB + k];
        else if (k < 2 * D_EMB) v = embed[(long)sC[i] * D_EMB + (k - D_EMB)];
        else                    v = embed[(long)nid   * D_EMB + (k - 2 * D_EMB)];
        sF[i][k] = (_Float16)v;
    }
    __syncthreads();

    // CDNA5 WMMA 16x16x32 f16 fragment indices (wave32)
    const int mrow = tid & 15;           // A row
    const int hi   = tid >> 4;           // lane half
    const int kg8  = hi * 8;             // A: K sub-group base
    const int ncol = tid & 15;           // B/C column

    v8f acc[4] = {};                     // 4 N-tiles of C (16x16 f32 each)

    for (int kk = 0; kk < K_TOT / 32; ++kk) {
        const int base = kk * 32;
        v16h a;
        #pragma unroll
        for (int t = 0; t < 8; ++t) a[t]     = sF[mrow][base + kg8 + t];       // K: base+{0..7|8..15}
        #pragma unroll
        for (int t = 0; t < 8; ++t) a[8 + t] = sF[mrow][base + 16 + kg8 + t];  // K: base+16+{...}

        const int kb = base + kg8;       // B: K rows base+{0..7|8..15} then base+16+{...}, matching A
        #pragma unroll
        for (int nt = 0; nt < 4; ++nt) {
            v16h b;
            #pragma unroll
            for (int t = 0; t < 16; ++t)
                b[t] = (_Float16)W1[(kb + (t >> 3) * 16 + (t & 7)) * HID + nt * 16 + ncol];
            acc[nt] = __builtin_amdgcn_wmma_f32_16x16x32_f16(
                false, a, false, b, (short)0, acc[nt], false, false);
        }
    }

    // bias + relu + dot with W2 (each lane holds 8 rows x 1 col per tile)
    float partial[8];
    #pragma unroll
    for (int r2 = 0; r2 < 8; ++r2) partial[r2] = 0.f;
    #pragma unroll
    for (int nt = 0; nt < 4; ++nt) {
        int n = nt * 16 + ncol;
        float w2n = W2[n];
        float b1n = b1[n];
        #pragma unroll
        for (int r2 = 0; r2 < 8; ++r2) {
            float hv = acc[nt][r2] + b1n;
            hv = hv > 0.f ? hv : 0.f;
            partial[r2] += hv * w2n;
        }
    }
    #pragma unroll
    for (int r2 = 0; r2 < 8; ++r2) sP[tid][r2] = partial[r2];
    __syncthreads();

    // Row m<8 lives in lanes 0..15 (elem m); row m>=8 in lanes 16..31 (elem m-8)
    if (tid < 16 && (m0 + tid) < count) {
        float s = b2[0];
        if (tid < 8) { for (int l = 0;  l < 16; ++l) s += sP[l][tid];     }
        else         { for (int l = 16; l < 32; ++l) s += sP[l][tid - 8]; }
        float gate = 1.0f / (1.0f + expf(-s));
        int r = sR[tid], c = sC[tid];
        if (r == nid) atomicAdd(&Gout[c], gate);   // mask[nid, c]
        if (c == nid) atomicAdd(&Gin[r],  gate);   // mask[r, nid]
    }
}

// ---------------- K3: logits[c] = sum_j w_j * (x[j] . Wg[:,c]) ----------------
__global__ void reduce_kernel(const float* __restrict__ x, const float* __restrict__ Wg,
                              const int* __restrict__ nodeid,
                              const float* __restrict__ A_cnt, const float* __restrict__ Gout,
                              const float* __restrict__ Gin, float* __restrict__ logits, int N) {
    int j = blockIdx.x * blockDim.x + threadIdx.x;
    if (j >= N) return;
    int nid = nodeid[0];
    float w = A_cnt[j] * 0.5f * (Gout[j] + Gin[j]);
    if (j == nid || w == 0.0f) return;             // ~16 survivors
    float acc[NCLS];
    #pragma unroll
    for (int c = 0; c < NCLS; ++c) acc[c] = 0.f;
    const float* xr = x + (long)j * 64;
    for (int k = 0; k < 64; ++k) {
        float xv = xr[k];
        #pragma unroll
        for (int c = 0; c < NCLS; ++c) acc[c] += xv * Wg[k * NCLS + c];
    }
    #pragma unroll
    for (int c = 0; c < NCLS; ++c) atomicAdd(&logits[c], w * acc[c]);
}

// ---------------- K4: softmax over 7 ----------------
__global__ void softmax_kernel(const float* __restrict__ logits, float* __restrict__ out) {
    if (threadIdx.x == 0) {
        float v[NCLS], mx = -1e30f;
        for (int c = 0; c < NCLS; ++c) { v[c] = logits[c]; mx = fmaxf(mx, v[c]); }
        float s = 0.f;
        for (int c = 0; c < NCLS; ++c) { v[c] = expf(v[c] - mx); s += v[c]; }
        for (int c = 0; c < NCLS; ++c) out[c] = v[c] / s;
    }
}

extern "C" void kernel_launch(void* const* d_in, const int* in_sizes, int n_in,
                              void* d_out, int out_size, void* d_ws, size_t ws_size,
                              hipStream_t stream) {
    (void)n_in; (void)out_size; (void)ws_size;
    const float* x      = (const float*)d_in[0];
    const float* embed  = (const float*)d_in[1];
    const int*   row    = (const int*)d_in[2];
    const int*   col    = (const int*)d_in[3];
    const float* adj    = (const float*)d_in[4];
    const float* W1     = (const float*)d_in[5];
    const float* b1     = (const float*)d_in[6];
    const float* W2     = (const float*)d_in[7];
    const float* b2     = (const float*)d_in[8];
    const float* Wg     = (const float*)d_in[9];
    const int*   nodeid = (const int*)d_in[10];

    const int N = in_sizes[1] / D_EMB;   // 8192
    const int E = in_sizes[2];           // 131072

    // workspace layout
    float* A_cnt    = (float*)d_ws;          // N
    float* Gout     = A_cnt + N;             // N
    float* Gin      = Gout + N;              // N
    float* logits   = Gin + N;               // 8 floats
    int*   counters = (int*)(logits + 8);    // 8 ints (counters[0] = incident count)
    int*   inc_list = counters + 8;          // E ints

    const int zero_words = 3 * N + 16;
    init_ws_kernel<<<(zero_words + 255) / 256, 256, 0, stream>>>((int*)d_ws, zero_words);
    scan_kernel<<<(E + 255) / 256, 256, 0, stream>>>(row, col, adj, nodeid,
                                                     A_cnt, counters, inc_list, E);
    mlp_gate_kernel<<<(E + 15) / 16, 32, 0, stream>>>(embed, row, col, W1, b1, W2, b2,
                                                      nodeid, counters, inc_list, Gout, Gin);
    reduce_kernel<<<(N + 255) / 256, 256, 0, stream>>>(x, Wg, nodeid, A_cnt, Gout, Gin,
                                                       logits, N);
    softmax_kernel<<<1, 32, 0, stream>>>(logits, (float*)d_out);
}
